// BlockSparseAttention_5007931867259
// MI455X (gfx1250) — matrix operand order repaired and checked by reading, not compiled, not run on hardware
//
#include <hip/hip_runtime.h>

// ---------------------------------------------------------------------------
// Block-sparse sliding-window attention for MI455X (gfx1250, wave32, WMMA).
// B=2, S=4096, H=16, D=64; BLOCK=64, WINDOW=256 -> 5 neighbor key blocks.
// One workgroup per (b, h, query-block); 4 waves, each owns 16 query rows.
// f16 WMMA (v_wmma_f32_16x16x32_f16) with fp32 accumulation; softmax in fp32.
// ---------------------------------------------------------------------------

typedef __attribute__((ext_vector_type(16))) _Float16 v16h;
typedef __attribute__((ext_vector_type(8)))  float    v8f;
typedef __attribute__((ext_vector_type(4)))  float    f32x4;

#define BQ   2
#define SQ   4096
#define HQ   16
#define DQ   64
#define NBLK (SQ / 64)   // 64 query/key blocks
#define HALFW 2          // window/2 in blocks
#define LSTR 72          // LDS stride (halves) for 64-wide f16 tiles (144B rows, 16B-aligned)
#define PSTR 320         // LDS stride (halves) for score/P buffer (640B rows)

union H4   { unsigned long long u; _Float16 h[4]; };
union F4H8 { f32x4 v; _Float16 h[8]; };

// Stage a 64x64 fp32 tile (row stride H*D floats in global) into LDS as f16.
// transpose==0: dst[row][col]  (stride LSTR)   -- used for Q and K
// transpose==1: dst[col][row]  (stride LSTR)   -- used for V (so B-operand K-dim is contiguous)
__device__ __forceinline__ void stage_tile(const float* __restrict__ src,
                                           _Float16* dst, int tid, int transpose) {
  #pragma unroll
  for (int i = 0; i < 8; ++i) {
    int idx = tid + i * 128;
    int r  = idx >> 4;       // row 0..63
    int c4 = idx & 15;       // float4 index within row
    f32x4 f = *(const f32x4*)(src + r * (HQ * DQ) + c4 * 4);
    if (transpose) {
      dst[(c4 * 4 + 0) * LSTR + r] = (_Float16)f.x;
      dst[(c4 * 4 + 1) * LSTR + r] = (_Float16)f.y;
      dst[(c4 * 4 + 2) * LSTR + r] = (_Float16)f.z;
      dst[(c4 * 4 + 3) * LSTR + r] = (_Float16)f.w;
    } else {
      H4 t;
      t.h[0] = (_Float16)f.x; t.h[1] = (_Float16)f.y;
      t.h[2] = (_Float16)f.z; t.h[3] = (_Float16)f.w;
      *(unsigned long long*)(dst + r * LSTR + c4 * 4) = t.u;
    }
  }
}

// A-operand (16x32 f16, MxK): lane m = lane&15, g = lane>>4.
// halves 0..7  = K[k0 + g*8 .. +7], halves 8..15 = K[k0 + 16 + g*8 .. +7].
__device__ __forceinline__ v16h load_a(const _Float16* buf, int stride,
                                       int row0, int k0, int lane) {
  int g = lane >> 4, m = lane & 15;
  const _Float16* p = buf + (row0 + m) * stride + k0 + g * 8;
  F4H8 u0, u1;
  u0.v = *(const f32x4*)(p);
  u1.v = *(const f32x4*)(p + 16);
  v16h a;
  #pragma unroll
  for (int i = 0; i < 8; ++i) { a[i] = u0.h[i]; a[i + 8] = u1.h[i]; }
  return a;
}

// B-operand (32x16 f16, KxN): lane n = lane&15, g = lane>>4.
// halves 0..7 = K[k0 + g*8 .. +7], halves 8..15 = K[k0 + 16 + g*8 .. +7] of column n (LDS row n0+n).
__device__ __forceinline__ v16h load_b(const _Float16* buf, int stride,
                                       int n0, int k0, int lane) {
  int g = lane >> 4, n = lane & 15;
  const _Float16* p = buf + (n0 + n) * stride + k0 + g * 8;
  F4H8 u0, u1;
  u0.v = *(const f32x4*)(p);
  u1.v = *(const f32x4*)(p + 16);
  v16h b;
  #pragma unroll
  for (int i = 0; i < 8; ++i) { b[i] = u0.h[i]; b[i + 8] = u1.h[i]; }
  return b;
}

__device__ __forceinline__ v8f wmma_f16(v16h a, v16h b, v8f c) {
  return __builtin_amdgcn_wmma_f32_16x16x32_f16(false, a, false, b,
                                                (short)0, c, false, false);
}

__global__ __launch_bounds__(128)
void bsattn_gfx1250_kernel(const float* __restrict__ q,
                           const float* __restrict__ k,
                           const float* __restrict__ v,
                           float* __restrict__ out) {
  __shared__ _Float16 Ql[64 * LSTR];   // Q tile, f16, rows=query, cols=d
  __shared__ _Float16 KV[64 * LSTR];   // shared K (rows=key,d) / V^T (rows=d,e) buffer
  __shared__ _Float16 SP[64 * PSTR];   // scores then P=exp(s-max), f16
  __shared__ float    rinv[64];        // 1/rowsum

  const int tid   = threadIdx.x;
  const int lane  = tid & 31;
  const int wave  = tid >> 5;
  const int mBase = wave * 16;         // this wave's query-row strip

  const int bid = blockIdx.x;
  const int n0  = bid % NBLK;
  const int h   = (bid / NBLK) % HQ;
  const int b   = bid / (NBLK * HQ);

  int jlo = n0 - HALFW; if (jlo < 0) jlo = 0;
  int jhi = n0 + HALFW; if (jhi > NBLK - 1) jhi = NBLK - 1;
  const int nv = jhi - jlo + 1;        // 3..5 valid key blocks

  // ---- stage Q -----------------------------------------------------------
  const float* qsrc = q + (((size_t)b * SQ + (size_t)n0 * 64) * HQ + h) * DQ;
  stage_tile(qsrc, Ql, tid, 0);
  __syncthreads();

  const v16h aq0 = load_a(Ql, LSTR, mBase, 0,  lane);
  const v16h aq1 = load_a(Ql, LSTR, mBase, 32, lane);

  // ---- phase 1: scores S = Q K^T into SP (raw f16) -----------------------
  for (int s = 0; s < nv; ++s) {
    const int j = jlo + s;
    __syncthreads();
    stage_tile(k + (((size_t)b * SQ + (size_t)j * 64) * HQ + h) * DQ, KV, tid, 0);
    __syncthreads();
    const int g = lane >> 4, n = lane & 15;
    #pragma unroll
    for (int t = 0; t < 4; ++t) {
      v8f c = {};
      c = wmma_f16(aq0, load_b(KV, LSTR, t * 16, 0,  lane), c);
      c = wmma_f16(aq1, load_b(KV, LSTR, t * 16, 32, lane), c);
      const int cb = s * 64 + t * 16 + n;
      #pragma unroll
      for (int r = 0; r < 8; ++r)
        SP[(mBase + g * 8 + r) * PSTR + cb] = (_Float16)c[r];
    }
  }

  // ---- phase 2: softmax over this wave's 16 rows -------------------------
  {
    const int row = lane & 15, hf = lane >> 4;
    const int m = mBase + row;
    const int cols = nv * 64, hc = cols >> 1, c0 = hf * hc;
    const float sc = 0.125f;             // 1/sqrt(64)
    float mx = -1e30f;
    for (int c = c0; c < c0 + hc; ++c) {
      float sv = (float)SP[m * PSTR + c];
      mx = fmaxf(mx, sv);
    }
    mx = fmaxf(mx, __shfl_xor(mx, 16, 32));
    mx *= sc;
    float sum = 0.0f;
    for (int c = c0; c < c0 + hc; ++c) {
      float sv = (float)SP[m * PSTR + c] * sc;
      float p = __expf(sv - mx);
      sum += p;
      SP[m * PSTR + c] = (_Float16)p;    // in-place: P = exp(s - max)
    }
    sum += __shfl_xor(sum, 16, 32);
    if (lane < 16) rinv[m] = 1.0f / sum;
  }

  // ---- phase 3: O = P V (unnormalized), V staged transposed --------------
  v8f acc[4] = {};
  for (int s = 0; s < nv; ++s) {
    const int j = jlo + s;
    __syncthreads();
    stage_tile(v + (((size_t)b * SQ + (size_t)j * 64) * HQ + h) * DQ, KV, tid, 1);
    __syncthreads();
    #pragma unroll
    for (int c = 0; c < 2; ++c) {
      const v16h ap = load_a(SP, PSTR, mBase, s * 64 + c * 32, lane);
      #pragma unroll
      for (int t = 0; t < 4; ++t)
        acc[t] = wmma_f16(ap, load_b(KV, LSTR, t * 16, c * 32, lane), acc[t]);
    }
  }

  // ---- write out, scaled by 1/rowsum -------------------------------------
  const int g = lane >> 4, n = lane & 15;
  float inv[8];
  #pragma unroll
  for (int r = 0; r < 8; ++r) inv[r] = rinv[mBase + g * 8 + r];
  #pragma unroll
  for (int t = 0; t < 4; ++t) {
    #pragma unroll
    for (int r = 0; r < 8; ++r) {
      const int m = mBase + g * 8 + r;
      const int d = t * 16 + n;
      out[(((size_t)b * SQ + (size_t)n0 * 64 + m) * HQ + h) * DQ + d] =
          acc[t][r] * inv[r];
    }
  }
}

extern "C" void kernel_launch(void* const* d_in, const int* in_sizes, int n_in,
                              void* d_out, int out_size, void* d_ws, size_t ws_size,
                              hipStream_t stream) {
  (void)in_sizes; (void)n_in; (void)out_size; (void)d_ws; (void)ws_size;
  const float* q = (const float*)d_in[0];
  const float* k = (const float*)d_in[1];
  const float* v = (const float*)d_in[2];
  float* out = (float*)d_out;
  dim3 grid(BQ * HQ * NBLK);   // 2048 workgroups: one per (b, h, query block)
  dim3 block(128);             // 4 waves of 32
  bsattn_gfx1250_kernel<<<grid, block, 0, stream>>>(q, k, v, out);
}
